// PocketEncoder_10642928959555
// MI455X (gfx1250) — matrix operand order repaired and checked by reading, not compiled, not run on hardware
//
#include <hip/hip_runtime.h>
#include <math.h>

// ---------------------------------------------------------------------------
// GVP-GNN pocket encoder for MI455X (gfx1250, wave32, WMMA bf16).
// All large matmuls go through v_wmma_f32_16x16x32_bf16. Node tensors
// (~20MB) live in L2 (192MB); scatter-mean uses native f32 global atomics.
// ---------------------------------------------------------------------------

typedef __attribute__((ext_vector_type(16))) __bf16 v16bf;
typedef __attribute__((ext_vector_type(8)))  __bf16 v8bf;
typedef __attribute__((ext_vector_type(8)))  float  v8f;

__device__ __forceinline__ float sigmoidf_(float x) { return 1.0f / (1.0f + __expf(-x)); }

// A-fragment (16x32 bf16) from an LDS tile, per CDNA5 WMMA layout:
// lane l: m = l&15, kb = (l>>4)*8; slots 0..7 -> K=k0+kb.., slots 8..15 -> K=k0+16+kb..
__device__ __forceinline__ v16bf load_a_frag(const __bf16* A, int lda, int lane, int k0) {
  const int m  = lane & 15;
  const int kb = (lane >> 4) << 3;
  const __bf16* row = A + m * lda + k0 + kb;
  v8bf lo = *(const v8bf*)(row);
  v8bf hi = *(const v8bf*)(row + 16);
  return __builtin_shufflevector(lo, hi, 0,1,2,3,4,5,6,7,8,9,10,11,12,13,14,15);
}

// B-fragment: weights pre-packed so each lane reads 16 contiguous bf16 (32B).
__device__ __forceinline__ v16bf load_b_frag(const __bf16* Bp, int kt, int nt, int lane) {
  return *(const v16bf*)(Bp + ((size_t)((kt * 16 + nt) * 32 + lane)) * 16);
}

// ---------------------------------------------------------------------------
// Pack fp32 (K,N) row-major weights into WMMA B-fragment order, bf16.
// B layout (32x16, 16-bit): lane l -> n = l&15; K slot j -> k = kt*32 + (l>>4)*8 + (j&7) + (j>>3)*16.
// ---------------------------------------------------------------------------
__global__ void pack_w_kernel(const float* __restrict__ W, __bf16* __restrict__ out,
                              int K, int N, int Kp) {
  const int ntiles = N >> 4;
  const int total  = (Kp >> 5) * ntiles * 512;
  for (int idx = blockIdx.x * blockDim.x + threadIdx.x; idx < total;
       idx += gridDim.x * blockDim.x) {
    int kt  = idx / (ntiles * 512);
    int rem = idx - kt * (ntiles * 512);
    int nt  = rem >> 9;
    int r2  = rem & 511;
    int ln  = r2 >> 4;
    int j   = r2 & 15;
    int k   = kt * 32 + ((ln >> 4) << 3) + (j & 7) + ((j >> 3) << 4);
    int n   = (nt << 4) + (ln & 15);
    out[idx] = (k < K) ? (__bf16)W[(size_t)k * N + n] : (__bf16)0.0f;
  }
}

// ---------------------------------------------------------------------------
// Fused edge-message kernel: gathers endpoints, runs the 3-layer GVP MLP
// (two relu GEMMs + one linear GEMM, all WMMA bf16), scatters with atomics.
// One wave per 16-edge tile, 2 waves per block.
//   K1P: padded K of GEMM1 (64 for conv0 [K1=51], 576 for conv1/2 [K1=551])
//   AC : LDS A-buffer columns = max(K1P, 288)  (A3 reuses the A1 buffer)
// ---------------------------------------------------------------------------
template <int K1P, int AC>
__launch_bounds__(64)
__global__ void edge_conv_kernel(
    const float* __restrict__ s_in, const float* __restrict__ v_in, int si,
    const int* __restrict__ src, const int* __restrict__ dst,
    const float* __restrict__ es, const float* __restrict__ ev,
    const float* __restrict__ wh1, const float* __restrict__ wv1,
    const float* __restrict__ wh2, const float* __restrict__ wv2,
    const float* __restrict__ wh3, const float* __restrict__ wv3,
    const __bf16* __restrict__ w1p, const float* __restrict__ b1,
    const __bf16* __restrict__ w2p, const float* __restrict__ b2,
    const __bf16* __restrict__ w3p, const float* __restrict__ b3,
    float* __restrict__ agg_s, float* __restrict__ agg_v, float* __restrict__ cnt,
    int nTiles)
{
  __shared__ __align__(16) __bf16 shA[2][16][AC];    // A1, later A3 ([s2, vn3] pad 288)
  __shared__ __align__(16) __bf16 shB[2][16][288];   // A2 = [s1, vn2] pad 288
  __shared__ int   shDst[2][16];
  __shared__ float shVo[2][16][12];                   // [0..9)=vo3, [9..12)=vn3

  const int lane = threadIdx.x & 31;
  const int wv   = threadIdx.x >> 5;
  const int tile = blockIdx.x * 2 + wv;
  if (tile >= nTiles) return;
  const int K1 = 2 * si + 32;         // concat scalar width (vn tail adds 7)
  const int eb = tile * 16;

  // ---- Phase 0: gather A1 rows + per-edge small vector math ----------------
  if (lane < 16) shDst[wv][lane] = dst[eb + lane];
  for (int m = 0; m < 16; ++m) {
    const int e = eb + m;
    const int sm = src[e], dm = dst[e];
    for (int c = lane; c < K1; c += 32) {
      float val;
      if (c < si)            val = s_in[(size_t)sm * si + c];
      else if (c < si + 32)  val = es[(size_t)e * 32 + (c - si)];
      else                   val = s_in[(size_t)dm * si + (c - si - 32)];
      shA[wv][m][c] = (__bf16)val;
    }
    for (int c = K1 + lane; c < K1P; c += 32) shA[wv][m][c] = (__bf16)0.0f; // pad (vn overwrites)
    for (int c = 259 + lane; c < 288; c += 32) shB[wv][m][c] = (__bf16)0.0f; // A2 pad
  }
  if (lane < 16) {
    const int m = lane, e = eb + m;
    const int sm = src[e], dm = dst[e];
    float vc[7][3];
    for (int c = 0; c < 3; ++c) {
      vc[0][c] = v_in[(size_t)sm * 9 + 0 + c];
      vc[1][c] = v_in[(size_t)sm * 9 + 3 + c];
      vc[2][c] = v_in[(size_t)sm * 9 + 6 + c];
      vc[3][c] = ev[(size_t)e * 3 + c];
      vc[4][c] = v_in[(size_t)dm * 9 + 0 + c];
      vc[5][c] = v_in[(size_t)dm * 9 + 3 + c];
      vc[6][c] = v_in[(size_t)dm * 9 + 6 + c];
    }
    // gvp0: vh = vcat @ wh1 (7x7); vn -> A1 tail; vo1 = vh @ wv1 (7x3), gated
    float vh[7][3];
    for (int h = 0; h < 7; ++h)
      for (int c = 0; c < 3; ++c) {
        float a = 0.f;
        for (int r = 0; r < 7; ++r) a += vc[r][c] * wh1[r * 7 + h];
        vh[h][c] = a;
      }
    for (int h = 0; h < 7; ++h) {
      float q = vh[h][0]*vh[h][0] + vh[h][1]*vh[h][1] + vh[h][2]*vh[h][2];
      shA[wv][m][K1 + h] = (__bf16)sqrtf(fmaxf(q, 1e-8f));
    }
    float vo[3][3];
    for (int o = 0; o < 3; ++o) {
      for (int c = 0; c < 3; ++c) {
        float a = 0.f;
        for (int h = 0; h < 7; ++h) a += vh[h][c] * wv1[h * 3 + o];
        vo[o][c] = a;
      }
      float q = vo[o][0]*vo[o][0] + vo[o][1]*vo[o][1] + vo[o][2]*vo[o][2];
      float g = sigmoidf_(sqrtf(fmaxf(q, 1e-8f)));
      vo[o][0] *= g; vo[o][1] *= g; vo[o][2] *= g;
    }
    // gvp1: vh2 = vo @ wh2 (3x3); vn2 -> A2 tail; vo2 = vh2 @ wv2, gated
    float vh2[3][3];
    for (int h = 0; h < 3; ++h)
      for (int c = 0; c < 3; ++c) {
        float a = 0.f;
        for (int r = 0; r < 3; ++r) a += vo[r][c] * wh2[r * 3 + h];
        vh2[h][c] = a;
      }
    for (int h = 0; h < 3; ++h) {
      float q = vh2[h][0]*vh2[h][0] + vh2[h][1]*vh2[h][1] + vh2[h][2]*vh2[h][2];
      shB[wv][m][256 + h] = (__bf16)sqrtf(fmaxf(q, 1e-8f));
    }
    float vo2[3][3];
    for (int o = 0; o < 3; ++o) {
      for (int c = 0; c < 3; ++c) {
        float a = 0.f;
        for (int h = 0; h < 3; ++h) a += vh2[h][c] * wv2[h * 3 + o];
        vo2[o][c] = a;
      }
      float q = vo2[o][0]*vo2[o][0] + vo2[o][1]*vo2[o][1] + vo2[o][2]*vo2[o][2];
      float g = sigmoidf_(sqrtf(fmaxf(q, 1e-8f)));
      vo2[o][0] *= g; vo2[o][1] *= g; vo2[o][2] *= g;
    }
    // gvp2: vh3 = vo2 @ wh3; vn3 stashed; vo3 = vh3 @ wv3 (no gate)
    float vh3[3][3];
    for (int h = 0; h < 3; ++h)
      for (int c = 0; c < 3; ++c) {
        float a = 0.f;
        for (int r = 0; r < 3; ++r) a += vo2[r][c] * wh3[r * 3 + h];
        vh3[h][c] = a;
      }
    for (int h = 0; h < 3; ++h) {
      float q = vh3[h][0]*vh3[h][0] + vh3[h][1]*vh3[h][1] + vh3[h][2]*vh3[h][2];
      shVo[wv][m][9 + h] = sqrtf(fmaxf(q, 1e-8f));
    }
    for (int o = 0; o < 3; ++o)
      for (int c = 0; c < 3; ++c) {
        float a = 0.f;
        for (int h = 0; h < 3; ++h) a += vh3[h][c] * wv3[h * 3 + o];
        shVo[wv][m][o * 3 + c] = a;
      }
  }
  __syncthreads();

  const int nc = lane & 15;
  const int mb = (lane >> 4) << 3;

  // ---- GEMM1: (16 x K1P) @ W1 -> relu -> A2 --------------------------------
  {
    constexpr int KT = K1P / 32;
    for (int nt = 0; nt < 16; ++nt) {
      v8f c = {0.f,0.f,0.f,0.f,0.f,0.f,0.f,0.f};
      for (int kt = 0; kt < KT; ++kt) {
        v16bf a = load_a_frag(&shA[wv][0][0], AC, lane, kt * 32);
        v16bf b = load_b_frag(w1p, kt, nt, lane);
        if (kt + 1 < KT)
          __builtin_prefetch(w1p + ((size_t)(((kt + 1) * 16 + nt) * 32 + lane)) * 16, 0, 0);
        c = __builtin_amdgcn_wmma_f32_16x16x32_bf16(false, a, false, b, (short)0, c, false, false);
      }
      float bias = b1[nt * 16 + nc];
      for (int r = 0; r < 8; ++r)
        shB[wv][mb + r][nt * 16 + nc] = (__bf16)fmaxf(c[r] + bias, 0.f);
    }
  }
  __syncthreads();

  // ---- GEMM2: (16 x 288) @ W2 -> relu -> A3 (reuses shA); add vn3+pad ------
  {
    for (int nt = 0; nt < 16; ++nt) {
      v8f c = {0.f,0.f,0.f,0.f,0.f,0.f,0.f,0.f};
      for (int kt = 0; kt < 9; ++kt) {
        v16bf a = load_a_frag(&shB[wv][0][0], 288, lane, kt * 32);
        v16bf b = load_b_frag(w2p, kt, nt, lane);
        c = __builtin_amdgcn_wmma_f32_16x16x32_bf16(false, a, false, b, (short)0, c, false, false);
      }
      float bias = b2[nt * 16 + nc];
      for (int r = 0; r < 8; ++r)
        shA[wv][mb + r][nt * 16 + nc] = (__bf16)fmaxf(c[r] + bias, 0.f);
    }
    if (lane < 16)
      for (int h = 0; h < 3; ++h)
        shA[wv][lane][256 + h] = (__bf16)shVo[wv][lane][9 + h];
    for (int idx = lane; idx < 16 * 29; idx += 32) {
      int mm = idx / 29, cc = 259 + idx % 29;
      shA[wv][mm][cc] = (__bf16)0.0f;
    }
  }
  __syncthreads();

  // ---- GEMM3: (16 x 288) @ W3 (linear) -> scatter-add ----------------------
  {
    for (int nt = 0; nt < 16; ++nt) {
      v8f c = {0.f,0.f,0.f,0.f,0.f,0.f,0.f,0.f};
      for (int kt = 0; kt < 9; ++kt) {
        v16bf a = load_a_frag(&shA[wv][0][0], AC, lane, kt * 32);
        v16bf b = load_b_frag(w3p, kt, nt, lane);
        c = __builtin_amdgcn_wmma_f32_16x16x32_bf16(false, a, false, b, (short)0, c, false, false);
      }
      float bias = b3[nt * 16 + nc];
      for (int r = 0; r < 8; ++r) {
        int dn = shDst[wv][mb + r];
        unsafeAtomicAdd(&agg_s[(size_t)dn * 256 + nt * 16 + nc], c[r] + bias);
      }
    }
    if (lane < 16) {
      int dn = shDst[wv][lane];
      unsafeAtomicAdd(&cnt[dn], 1.0f);
      for (int i = 0; i < 9; ++i)
        unsafeAtomicAdd(&agg_v[(size_t)dn * 9 + i], shVo[wv][lane][i]);
    }
  }
}

// ---------------------------------------------------------------------------
// Fused feed-forward (GVP ff0+ff1) per 16-node tile, one wave per block.
// GEMM1: K=288 (262 real) -> N=1024 relu; GEMM2: K=1056 (1030) -> N=256.
// Writes ds/dv (residual terms) directly, no atomics.
// ---------------------------------------------------------------------------
__launch_bounds__(32)
__global__ void node_ff_kernel(
    const float* __restrict__ s, const float* __restrict__ v,
    const float* __restrict__ wh0, const float* __restrict__ wv0,
    const float* __restrict__ wh1f, const float* __restrict__ wv1f,
    const __bf16* __restrict__ w0p, const float* __restrict__ b0,
    const __bf16* __restrict__ w1p, const float* __restrict__ b1,
    float* __restrict__ ds_out, float* __restrict__ dv_out, int nTiles)
{
  __shared__ __align__(16) __bf16 shA[16][288];     // [s(256), vn(6)] pad 288
  __shared__ __align__(16) __bf16 shH[16][1056];    // [h(1024), vn2(6)] pad 1056
  __shared__ float shDv[16][9];

  const int lane = threadIdx.x;
  const int tile = blockIdx.x;
  if (tile >= nTiles) return;
  const int nb = tile * 16;

  for (int m = 0; m < 16; ++m) {
    const size_t node = (size_t)(nb + m);
    for (int c = lane; c < 256; c += 32) shA[m][c] = (__bf16)s[node * 256 + c];
    for (int c = 262 + lane; c < 288; c += 32) shA[m][c] = (__bf16)0.0f;
    for (int c = 1030 + lane; c < 1056; c += 32) shH[m][c] = (__bf16)0.0f;
  }
  if (lane < 16) {
    const int m = lane;
    const size_t node = (size_t)(nb + m);
    float vr[3][3];
    for (int r = 0; r < 3; ++r)
      for (int c = 0; c < 3; ++c) vr[r][c] = v[node * 9 + r * 3 + c];
    float vh[6][3];
    for (int h = 0; h < 6; ++h)
      for (int c = 0; c < 3; ++c) {
        float a = 0.f;
        for (int r = 0; r < 3; ++r) a += vr[r][c] * wh0[r * 6 + h];
        vh[h][c] = a;
      }
    for (int h = 0; h < 6; ++h) {
      float q = vh[h][0]*vh[h][0] + vh[h][1]*vh[h][1] + vh[h][2]*vh[h][2];
      shA[m][256 + h] = (__bf16)sqrtf(fmaxf(q, 1e-8f));
    }
    float vo[6][3];
    for (int o = 0; o < 6; ++o) {
      for (int c = 0; c < 3; ++c) {
        float a = 0.f;
        for (int h = 0; h < 6; ++h) a += vh[h][c] * wv0[h * 6 + o];
        vo[o][c] = a;
      }
      float q = vo[o][0]*vo[o][0] + vo[o][1]*vo[o][1] + vo[o][2]*vo[o][2];
      float g = sigmoidf_(sqrtf(fmaxf(q, 1e-8f)));
      vo[o][0] *= g; vo[o][1] *= g; vo[o][2] *= g;
    }
    float vh2[6][3];
    for (int h = 0; h < 6; ++h)
      for (int c = 0; c < 3; ++c) {
        float a = 0.f;
        for (int r = 0; r < 6; ++r) a += vo[r][c] * wh1f[r * 6 + h];
        vh2[h][c] = a;
      }
    for (int h = 0; h < 6; ++h) {
      float q = vh2[h][0]*vh2[h][0] + vh2[h][1]*vh2[h][1] + vh2[h][2]*vh2[h][2];
      shH[m][1024 + h] = (__bf16)sqrtf(fmaxf(q, 1e-8f));
    }
    for (int o = 0; o < 3; ++o)
      for (int c = 0; c < 3; ++c) {
        float a = 0.f;
        for (int h = 0; h < 6; ++h) a += vh2[h][c] * wv1f[h * 3 + o];
        shDv[m][o * 3 + c] = a;
      }
  }
  __syncthreads();

  const int nc = lane & 15;
  const int mb = (lane >> 4) << 3;

  for (int nt = 0; nt < 64; ++nt) {
    v8f c = {0.f,0.f,0.f,0.f,0.f,0.f,0.f,0.f};
    for (int kt = 0; kt < 9; ++kt) {
      v16bf a = load_a_frag(&shA[0][0], 288, lane, kt * 32);
      v16bf b = load_b_frag(w0p, kt, nt, lane);
      c = __builtin_amdgcn_wmma_f32_16x16x32_bf16(false, a, false, b, (short)0, c, false, false);
    }
    float bias = b0[nt * 16 + nc];
    for (int r = 0; r < 8; ++r)
      shH[mb + r][nt * 16 + nc] = (__bf16)fmaxf(c[r] + bias, 0.f);
  }
  __syncthreads();

  for (int nt = 0; nt < 16; ++nt) {
    v8f c = {0.f,0.f,0.f,0.f,0.f,0.f,0.f,0.f};
    for (int kt = 0; kt < 33; ++kt) {
      v16bf a = load_a_frag(&shH[0][0], 1056, lane, kt * 32);
      v16bf b = load_b_frag(w1p, kt, nt, lane);
      if (kt + 1 < 33)
        __builtin_prefetch(w1p + ((size_t)(((kt + 1) * 16 + nt) * 32 + lane)) * 16, 0, 0);
      c = __builtin_amdgcn_wmma_f32_16x16x32_bf16(false, a, false, b, (short)0, c, false, false);
    }
    float bias = b1[nt * 16 + nc];
    for (int r = 0; r < 8; ++r)
      ds_out[(size_t)(nb + mb + r) * 256 + nt * 16 + nc] = c[r] + bias;
  }
  if (lane < 16)
    for (int i = 0; i < 9; ++i)
      dv_out[(size_t)(nb + lane) * 9 + i] = shDv[lane][i];
}

// ---------------------------------------------------------------------------
// LayerNorm / vector-RMS normalize, one wave per node (8 nodes per block).
// x = agg ? ((res ? s : 0) + agg * (cnt ? 1/max(cnt,1) : 1)) : s ; same for v.
// ---------------------------------------------------------------------------
__launch_bounds__(256)
__global__ void node_norm_kernel(
    float* __restrict__ s, float* __restrict__ v,
    const float* __restrict__ agg_s, const float* __restrict__ agg_v,
    const float* __restrict__ cnt,
    const float* __restrict__ gamma, const float* __restrict__ beta,
    int n_nodes, int use_res)
{
  const int lane = threadIdx.x & 31;
  const int node = blockIdx.x * 8 + (threadIdx.x >> 5);
  if (node >= n_nodes) return;
  const size_t sb = (size_t)node * 256;
  float invc = 1.0f;
  if (cnt) invc = 1.0f / fmaxf(cnt[node], 1.0f);

  float x[8], sum = 0.f;
  for (int i = 0; i < 8; ++i) {
    int c = i * 32 + lane;
    float val = agg_s ? ((use_res ? s[sb + c] : 0.f) + agg_s[sb + c] * invc) : s[sb + c];
    x[i] = val; sum += val;
  }
  for (int o = 16; o > 0; o >>= 1) sum += __shfl_xor(sum, o, 32);
  const float mu = sum * (1.0f / 256.0f);
  float s2 = 0.f;
  for (int i = 0; i < 8; ++i) { float d = x[i] - mu; s2 += d * d; }
  for (int o = 16; o > 0; o >>= 1) s2 += __shfl_xor(s2, o, 32);
  const float rstd = rsqrtf(s2 * (1.0f / 256.0f) + 1e-5f);
  for (int i = 0; i < 8; ++i) {
    int c = i * 32 + lane;
    s[sb + c] = (x[i] - mu) * rstd * gamma[c] + beta[c];
  }

  const size_t vb = (size_t)node * 9;
  float vv[9], q = 0.f;
  for (int r = 0; r < 3; ++r) {
    float qr = 0.f;
    for (int c = 0; c < 3; ++c) {
      float val = agg_v ? ((use_res ? v[vb + r * 3 + c] : 0.f) + agg_v[vb + r * 3 + c] * invc)
                        : v[vb + r * 3 + c];
      vv[r * 3 + c] = val; qr += val * val;
    }
    q += fmaxf(qr, 1e-8f);
  }
  const float rms = sqrtf(q * (1.0f / 3.0f));
  if (lane < 9) v[vb + lane] = vv[lane] / rms;
}

__launch_bounds__(256)
__global__ void mean_kernel(const float* __restrict__ s, float* __restrict__ hid, int n_nodes) {
  const int c = threadIdx.x;
  const int chunk = (n_nodes + gridDim.x - 1) / gridDim.x;
  const int n0 = blockIdx.x * chunk;
  const int n1 = (n0 + chunk < n_nodes) ? n0 + chunk : n_nodes;
  float acc = 0.f;
  for (int n = n0; n < n1; ++n) acc += s[(size_t)n * 256 + c];
  unsafeAtomicAdd(&hid[c], acc * (1.0f / (float)n_nodes));
}

__launch_bounds__(256)
__global__ void head_kernel(const float* __restrict__ hid, const float* __restrict__ W,
                            const float* __restrict__ b, float* __restrict__ out) {
  const int j = threadIdx.x;
  if (j >= 250) return;
  float acc = b[j];
  for (int c = 0; c < 256; ++c) acc += hid[c] * W[(size_t)c * 250 + j];
  out[j] = acc;
}

// ---------------------------------------------------------------------------
// Host launcher. Input flattening (setup_inputs insertion order, depth-first):
//  0 nodes_s  1 nodes_v  2 edge_index  3 edges_s  4 edges_v
//  5..16  conv0.{gvp0,gvp1,gvp2}.{wh,ws_w,ws_b,wv}
// 17,18   ln.{gamma,beta}
// 19..42  conv1: conv.{gvp0,gvp1,gvp2}{..}, norm0{g,b}, ff0{..}, ff1{..}, norm1{g,b}
// 43..66  conv2 (same layout)
// 67,68   mean_w, mean_b
// ---------------------------------------------------------------------------
extern "C" void kernel_launch(void* const* d_in, const int* in_sizes, int n_in,
                              void* d_out, int out_size, void* d_ws, size_t ws_size,
                              hipStream_t stream) {
  const float* nodes_s = (const float*)d_in[0];
  const float* nodes_v = (const float*)d_in[1];
  const int*   eidx    = (const int*)d_in[2];
  const float* edges_s = (const float*)d_in[3];
  const float* edges_v = (const float*)d_in[4];
  const int N = in_sizes[0] / 6;
  const int E = in_sizes[2] / 2;
  const int* src = eidx;
  const int* dst = eidx + E;
  auto F = [&](int i) { return (const float*)d_in[i]; };

  char* ws = (char*)d_ws;
  size_t off = 0;
  auto take = [&](size_t bytes) -> char* {
    char* p = ws + off; off = (off + bytes + 255) & ~(size_t)255; return p;
  };
  float* s_cur = (float*)take((size_t)N * 256 * 4);
  float* v_cur = (float*)take((size_t)N * 9 * 4);
  float* agg_s = (float*)take((size_t)N * 256 * 4);
  float* agg_v = (float*)take((size_t)N * 9 * 4);
  float* cnt   = (float*)take((size_t)N * 4);
  float* hid   = (float*)take(256 * 4);
  auto takebf = [&](size_t elems) { return (__bf16*)take(elems * 2); };
  __bf16* c0w1 = takebf((size_t)64 * 256);
  __bf16* c0w2 = takebf((size_t)288 * 256);
  __bf16* c0w3 = takebf((size_t)288 * 256);
  __bf16* c1w1 = takebf((size_t)576 * 256);
  __bf16* c1w2 = takebf((size_t)288 * 256);
  __bf16* c1w3 = takebf((size_t)288 * 256);
  __bf16* c1f0 = takebf((size_t)288 * 1024);
  __bf16* c1f1 = takebf((size_t)1056 * 256);
  __bf16* c2w1 = takebf((size_t)576 * 256);
  __bf16* c2w2 = takebf((size_t)288 * 256);
  __bf16* c2w3 = takebf((size_t)288 * 256);
  __bf16* c2f0 = takebf((size_t)288 * 1024);
  __bf16* c2f1 = takebf((size_t)1056 * 256);

  auto pack = [&](int widx, __bf16* dstp, int K, int Nn, int Kp) {
    int total = (Kp / 32) * (Nn / 16) * 512;
    pack_w_kernel<<<dim3((total + 255) / 256), dim3(256), 0, stream>>>(F(widx), dstp, K, Nn, Kp);
  };
  pack(6,  c0w1,  51,  256,  64);
  pack(10, c0w2, 259,  256, 288);
  pack(14, c0w3, 259,  256, 288);
  pack(20, c1w1, 551,  256, 576);
  pack(24, c1w2, 259,  256, 288);
  pack(28, c1w3, 259,  256, 288);
  pack(34, c1f0, 262, 1024, 288);
  pack(38, c1f1, 1030, 256, 1056);
  pack(44, c2w1, 551,  256, 576);
  pack(48, c2w2, 259,  256, 288);
  pack(52, c2w3, 259,  256, 288);
  pack(58, c2f0, 262, 1024, 288);
  pack(62, c2f1, 1030, 256, 1056);

  const int eTiles  = (E + 15) / 16;
  const int eBlocks = (eTiles + 1) / 2;
  const int nTiles  = (N + 15) / 16;
  const int nBlocks = (N + 7) / 8;

  auto zero_agg = [&]() {
    hipMemsetAsync(agg_s, 0, (size_t)N * 256 * 4, stream);
    hipMemsetAsync(agg_v, 0, (size_t)N * 9 * 4, stream);
    hipMemsetAsync(cnt,   0, (size_t)N * 4, stream);
  };
  auto norm = [&](const float* g, const float* b, const float* as, const float* av,
                  const float* ct, int res) {
    node_norm_kernel<<<dim3(nBlocks), dim3(256), 0, stream>>>(s_cur, v_cur, as, av, ct, g, b, N, res);
  };

  // conv0 (si=6) + shared LN
  zero_agg();
  edge_conv_kernel<64, 288><<<dim3(eBlocks), dim3(64), 0, stream>>>(
      nodes_s, nodes_v, 6, src, dst, edges_s, edges_v,
      F(5), F(8), F(9), F(12), F(13), F(16),
      c0w1, F(7), c0w2, F(11), c0w3, F(15),
      agg_s, agg_v, cnt, eTiles);
  norm(F(17), F(18), agg_s, agg_v, cnt, 0);

  auto conv_layer = [&](int base, __bf16* w1, __bf16* w2, __bf16* w3, __bf16* f0, __bf16* f1) {
    zero_agg();
    edge_conv_kernel<576, 576><<<dim3(eBlocks), dim3(64), 0, stream>>>(
        s_cur, v_cur, 256, src, dst, edges_s, edges_v,
        F(base + 0), F(base + 3), F(base + 4), F(base + 7), F(base + 8), F(base + 11),
        w1, F(base + 2), w2, F(base + 6), w3, F(base + 10),
        agg_s, agg_v, cnt, eTiles);
    norm(F(base + 12), F(base + 13), agg_s, agg_v, cnt, 1);      // norm0(s+ds)
    node_ff_kernel<<<dim3(nTiles), dim3(32), 0, stream>>>(
        s_cur, v_cur,
        F(base + 14), F(base + 17), F(base + 18), F(base + 21),
        f0, F(base + 16), f1, F(base + 20),
        agg_s, agg_v, nTiles);
    norm(F(base + 22), F(base + 23), agg_s, agg_v, nullptr, 1);  // norm1(s+ff)
    norm(F(17), F(18), nullptr, nullptr, nullptr, 0);            // shared ln
  };
  conv_layer(19, c1w1, c1w2, c1w3, c1f0, c1f1);
  conv_layer(43, c2w1, c2w2, c2w3, c2f0, c2f1);

  hipMemsetAsync(hid, 0, 256 * 4, stream);
  mean_kernel<<<dim3(40), dim3(256), 0, stream>>>(s_cur, hid, N);
  head_kernel<<<dim3(1), dim3(256), 0, stream>>>(hid, F(67), F(68), (float*)d_out);
}
